// RationalBasicBlock_26654567039143
// MI455X (gfx1250) — matrix operand order repaired and checked by reading, not compiled, not run on hardware
//
#include <hip/hip_runtime.h>
#include <hip/hip_bf16.h>
#include <stdint.h>

typedef __attribute__((ext_vector_type(16))) __bf16       bf16x16;
typedef __attribute__((ext_vector_type(8)))  float        f32x8;
typedef __attribute__((ext_vector_type(4)))  unsigned int u32x4;

#define BB    64
#define CIN   96
#define COUT  192
#define HIN   56
#define HOUT  28
#define MTOT  (BB*HOUT*HOUT)   /* 50176 */
#define BM    128
#define BN    64
#define ASTR  104               /* 96 + 8 pad, keeps 16B alignment */

/* ---- workspace layout (bytes) ---- */
#define XB_ELTS  (BB*HIN*HIN*CIN)       /* 19267584 */
#define T1_ELTS  (BB*HOUT*HOUT*COUT)    /*  9633792 */
#define OFF_XB   ((size_t)0)
#define OFF_T1   (OFF_XB  + (size_t)XB_ELTS*2)
#define OFF_SC   (OFF_T1  + (size_t)T1_ELTS*2)
#define OFF_W1T  (OFF_SC  + (size_t)T1_ELTS*4)
#define OFF_W2T  (OFF_W1T + (size_t)(9*COUT*CIN)*2)
#define OFF_WST  (OFF_W2T + (size_t)(9*COUT*COUT)*2)
#define OFF_BN   (OFF_WST + (size_t)(COUT*CIN)*2)
#define OFF_COEF (OFF_BN  + (size_t)(6*COUT)*4)

/* ================= device helpers ================= */

static __device__ __forceinline__ f32x8 wmma_bf16(bf16x16 a, bf16x16 b, f32x8 c) {
  return __builtin_amdgcn_wmma_f32_16x16x32_bf16(false, a, false, b, (short)0, c, false, false);
}

/* async DMA: 16 bytes global -> LDS, tracked by ASYNCcnt (bypasses VGPRs) */
static __device__ __forceinline__ void async_ld16(void* lds_dst, const void* base, int byte_off) {
  const uint32_t lds_addr = (uint32_t)(uintptr_t)lds_dst;   /* low 32b of generic = LDS offset */
  asm volatile("global_load_async_to_lds_b128 %0, %1, %2"
               :: "v"(lds_addr), "v"(byte_off), "s"(base)
               : "memory");
}
static __device__ __forceinline__ void async_wait0() {
  asm volatile("s_wait_asynccnt 0" ::: "memory");
}

/* A fragment (16x32 bf16): lane<16 holds K {0..7,16..23}, lane>=16 holds {8..15,24..31} of row lane&15 */
static __device__ __forceinline__ bf16x16 lds_afrag(const __bf16* row, int kc, int lane) {
  const int hi = (lane & 16) ? 8 : 0;
  union { bf16x16 v; u32x4 q[2]; } u;
  u.q[0] = *(const u32x4*)(row + kc + hi);
  u.q[1] = *(const u32x4*)(row + kc + 16 + hi);
  return u.v;
}
/* B fragment (32x16 bf16), LDS stored column-major: lane<16 holds K {0..7,16..23}, lane>=16 K {8..15,24..31} of col lane&15 */
static __device__ __forceinline__ bf16x16 lds_bfrag(const __bf16* col, int kc, int lane) {
  const int ko = (lane & 16) ? 8 : 0;
  union { bf16x16 v; u32x4 q[2]; } u;
  u.q[0] = *(const u32x4*)(col + kc + ko);
  u.q[1] = *(const u32x4*)(col + kc + 16 + ko);
  return u.v;
}

/* rational activation, version A, degrees (5,4); c[0..5]=num, c[6..9]=|den| */
static __device__ __forceinline__ float rational_f(float x, const float* c) {
  float p = c[5];
#pragma unroll
  for (int i = 4; i >= 0; --i) p = __builtin_fmaf(p, x, c[i]);
  const float xa = __builtin_fabsf(x);
  float q = 1.0f, pw = xa;
#pragma unroll
  for (int j = 0; j < 4; ++j) { q = __builtin_fmaf(c[6 + j], pw, q); pw *= xa; }
  return p / q;
}

/* ================= prep kernels ================= */

__global__ void prep_params(const float* g1, const float* b1, const float* m1, const float* v1,
                            const float* g2, const float* b2, const float* m2, const float* v2,
                            const float* gs, const float* bs, const float* ms, const float* vs,
                            const float* nr, const float* dr, const float* ng, const float* dg,
                            const float* nb, const float* db,
                            float* bnbuf, float* coef) {
  const int t = threadIdx.x;
  if (t < COUT) {
    float s;
    s = g1[t] * rsqrtf(v1[t] + 1e-5f); bnbuf[t]           = s; bnbuf[COUT   + t] = b1[t] - m1[t] * s;
    s = g2[t] * rsqrtf(v2[t] + 1e-5f); bnbuf[2*COUT + t]  = s; bnbuf[3*COUT + t] = b2[t] - m2[t] * s;
    s = gs[t] * rsqrtf(vs[t] + 1e-5f); bnbuf[4*COUT + t]  = s; bnbuf[5*COUT + t] = bs[t] - ms[t] * s;
  }
  if (t < 36) {
    const float* nums[3] = { nr, ng, nb };
    const float* dens[3] = { dr, dg, db };
    const int v = t / 12, j = t % 12;
    float val = 0.0f;
    if (j < 6) val = nums[v][j];
    else if (j < 10) val = __builtin_fabsf(dens[v][j - 6]);
    coef[t] = val;
  }
}

/* NCHW fp32 -> NHWC bf16 via LDS transpose (both global sides coalesced) */
__global__ __launch_bounds__(256) void prep_x(const float* __restrict__ x, __bf16* __restrict__ xb) {
  __shared__ __bf16 tile[CIN][72];
  const int hw0 = blockIdx.x * 64;
  const int b   = blockIdx.y;
  const int tid = threadIdx.x;
  for (int t = tid; t < CIN * 64; t += 256) {
    const int c = t >> 6, j = t & 63;
    tile[c][j] = (__bf16)x[(b * CIN + c) * (HIN * HIN) + hw0 + j];
  }
  __syncthreads();
  for (int t = tid; t < 64 * CIN; t += 256) {
    const int hw = t / CIN, c = t % CIN;
    xb[(b * (HIN * HIN) + hw0 + hw) * CIN + c] = tile[c][hw];
  }
}

/* OIHW fp32 -> [tap][cout][cin] bf16 */
__global__ void prep_w1(const float* __restrict__ w, __bf16* __restrict__ o) {
  const int i = blockIdx.x * 256 + threadIdx.x;
  if (i < 9 * COUT * CIN) {
    const int co = (i % (COUT * CIN)) / CIN, ci = i % CIN, tap = i / (COUT * CIN);
    o[i] = (__bf16)w[(co * CIN + ci) * 9 + tap];
  }
}
__global__ void prep_w2(const float* __restrict__ w, __bf16* __restrict__ o) {
  const int i = blockIdx.x * 256 + threadIdx.x;
  if (i < 9 * COUT * COUT) {
    const int co = (i % (COUT * COUT)) / COUT, ci = i % COUT, tap = i / (COUT * COUT);
    o[i] = (__bf16)w[(co * COUT + ci) * 9 + tap];
  }
}
__global__ void prep_wsk(const float* __restrict__ w, __bf16* __restrict__ o) {
  const int i = blockIdx.x * 256 + threadIdx.x;
  if (i < COUT * CIN) o[i] = (__bf16)w[i];
}

/* ================= conv1 (3x3 s2) + fused 1x1 s2 shortcut =================
   Double-buffered LDS tiles filled by GLOBAL_LOAD_ASYNC_TO_LDS_B128:
   each tap issues the NEXT tile's DMA right after the barrier, then computes
   the current tile, hiding the HBM latency behind the WMMA stream.         */

__global__ __launch_bounds__(256) void conv1_kernel(
    const __bf16* __restrict__ xb, const __bf16* __restrict__ w1t,
    const __bf16* __restrict__ wst, const float* __restrict__ bnbuf,
    const float* __restrict__ coef, __bf16* __restrict__ t1, float* __restrict__ sc) {
  __shared__ __align__(16) __bf16 As[2][BM][ASTR];
  __shared__ __align__(16) __bf16 Bs[2][BN][ASTR];
  __shared__ __align__(16) __bf16 Bss[BN][ASTR];

  const int tid  = threadIdx.x;
  const int lane = tid & 31;
  const int wave = tid >> 5;
  const int wm = wave >> 1;          /* 0..3 -> 32 rows each  */
  const int wn = wave & 1;           /* 0..1 -> 32 cols each  */
  const int m0 = blockIdx.x * BM;
  const int n0 = blockIdx.y * BN;

  const f32x8 z = {0,0,0,0,0,0,0,0};
  f32x8 acc[2][2], sacc[2][2];
#pragma unroll
  for (int i = 0; i < 2; ++i)
#pragma unroll
    for (int j = 0; j < 2; ++j) { acc[i][j] = z; sacc[i][j] = z; }

  /* ---- tile issue helpers (async DMA into LDS) ---- */
  auto issueA = [&](int tap, int bufi) {
    const int kh = tap / 3, kw = tap % 3;
    for (int t = tid; t < BM * 12; t += 256) {
      const int r = t / 12, ch = t % 12;
      const int m = m0 + r;
      const int b = m / 784, rem = m % 784;
      const int oh = rem / 28, ow = rem % 28;
      const int ih = 2 * oh + kh - 1, iw = 2 * ow + kw - 1;
      __bf16* dst = &As[bufi][r][ch * 8];
      if (ih >= 0 && ih < HIN && iw >= 0 && iw < HIN) {
        const int off = (((b * HIN + ih) * HIN + iw) * CIN + ch * 8) * 2;
        async_ld16(dst, xb, off);
      } else {
        const u32x4 zz = {0, 0, 0, 0};
        *(u32x4*)dst = zz;                      /* pad row: zero-fill via DS */
      }
    }
  };
  auto issueB = [&](int tap, int bufi) {
    for (int t = tid; t < BN * 12; t += 256) {
      const int n = t / 12, ch = t % 12;
      const int off = ((tap * COUT + n0 + n) * CIN + ch * 8) * 2;
      async_ld16(&Bs[bufi][n][ch * 8], w1t, off);
    }
  };

  /* shortcut weights: async, once */
  for (int t = tid; t < BN * 12; t += 256) {
    const int n = t / 12, ch = t % 12;
    async_ld16(&Bss[n][ch * 8], wst, ((n0 + n) * CIN + ch * 8) * 2);
  }
  issueA(0, 0);
  issueB(0, 0);

  const int rowA = wm * 32 + (lane & 15);
  const int colB = wn * 32 + (lane & 15);
  int buf = 0;

  for (int tap = 0; tap < 9; ++tap) {
    async_wait0();         /* my DMA for tile 'tap' (and Bss) has landed */
    __syncthreads();       /* everyone's landed; everyone done reading buf^1 */
    if (tap < 8) { issueA(tap + 1, buf ^ 1); issueB(tap + 1, buf ^ 1); }

#pragma unroll
    for (int kc = 0; kc < 96; kc += 32) {
      const bf16x16 a0 = lds_afrag(&As[buf][rowA][0],      kc, lane);
      const bf16x16 a1 = lds_afrag(&As[buf][rowA + 16][0], kc, lane);
      const bf16x16 b0 = lds_bfrag(&Bs[buf][colB][0],      kc, lane);
      const bf16x16 b1 = lds_bfrag(&Bs[buf][colB + 16][0], kc, lane);
      acc[0][0] = wmma_bf16(a0, b0, acc[0][0]);
      acc[0][1] = wmma_bf16(a0, b1, acc[0][1]);
      acc[1][0] = wmma_bf16(a1, b0, acc[1][0]);
      acc[1][1] = wmma_bf16(a1, b1, acc[1][1]);
      if (tap == 4) {   /* center tap pixel == shortcut's 1x1 s2 input */
        const bf16x16 s0 = lds_bfrag(&Bss[colB][0],      kc, lane);
        const bf16x16 s1 = lds_bfrag(&Bss[colB + 16][0], kc, lane);
        sacc[0][0] = wmma_bf16(a0, s0, sacc[0][0]);
        sacc[0][1] = wmma_bf16(a0, s1, sacc[0][1]);
        sacc[1][0] = wmma_bf16(a1, s0, sacc[1][0]);
        sacc[1][1] = wmma_bf16(a1, s1, sacc[1][1]);
      }
    }
    buf ^= 1;
  }

  /* epilogue: BN1 + rational -> t1 (bf16 NHWC); BNs -> sc (fp32) */
  const float* bn1s = bnbuf;
  const float* bn1b = bnbuf + COUT;
  const float* bnss = bnbuf + 4 * COUT;
  const float* bnsb = bnbuf + 5 * COUT;
  const int mhi = (lane & 16) ? 8 : 0;
#pragma unroll
  for (int j = 0; j < 2; ++j) {
    const int n = n0 + wn * 32 + j * 16 + (lane & 15);
    const float s1 = bn1s[n], o1 = bn1b[n];
    const float ss = bnss[n], os = bnsb[n];
    const float* cf = coef + (n % 3) * 12;
    float c[10];
#pragma unroll
    for (int q = 0; q < 10; ++q) c[q] = cf[q];
#pragma unroll
    for (int i = 0; i < 2; ++i) {
#pragma unroll
      for (int r = 0; r < 8; ++r) {
        const int m = m0 + wm * 32 + i * 16 + r + mhi;
        const float y = __builtin_fmaf(acc[i][j][r], s1, o1);
        t1[m * COUT + n] = (__bf16)rational_f(y, c);
        sc[m * COUT + n] = __builtin_fmaf(sacc[i][j][r], ss, os);
      }
    }
  }
}

/* ================= conv2 (3x3 s1) + BN2 + shortcut add + rational ================= */

__global__ __launch_bounds__(256) void conv2_kernel(
    const __bf16* __restrict__ t1, const __bf16* __restrict__ w2t,
    const float* __restrict__ sc, const float* __restrict__ bnbuf,
    const float* __restrict__ coef, float* __restrict__ out) {
  __shared__ __align__(16) __bf16 As[2][BM][ASTR];
  __shared__ __align__(16) __bf16 Bs[2][BN][ASTR];

  const int tid  = threadIdx.x;
  const int lane = tid & 31;
  const int wave = tid >> 5;
  const int wm = wave >> 1;
  const int wn = wave & 1;
  const int m0 = blockIdx.x * BM;
  const int n0 = blockIdx.y * BN;

  const f32x8 z = {0,0,0,0,0,0,0,0};
  f32x8 acc[2][2];
#pragma unroll
  for (int i = 0; i < 2; ++i)
#pragma unroll
    for (int j = 0; j < 2; ++j) acc[i][j] = z;

  /* 18 chunks: 9 taps x 2 halves of Cin=192 */
  auto issueA = [&](int tt, int bufi) {
    const int tap = tt >> 1, half = tt & 1;
    const int kh = tap / 3, kw = tap % 3;
    for (int t = tid; t < BM * 12; t += 256) {
      const int r = t / 12, ch = t % 12;
      const int m = m0 + r;
      const int b = m / 784, rem = m % 784;
      const int oh = rem / 28, ow = rem % 28;
      const int ih = oh + kh - 1, iw = ow + kw - 1;
      __bf16* dst = &As[bufi][r][ch * 8];
      if (ih >= 0 && ih < HOUT && iw >= 0 && iw < HOUT) {
        const int off = (((b * HOUT + ih) * HOUT + iw) * COUT + half * 96 + ch * 8) * 2;
        async_ld16(dst, t1, off);
      } else {
        const u32x4 zz = {0, 0, 0, 0};
        *(u32x4*)dst = zz;
      }
    }
  };
  auto issueB = [&](int tt, int bufi) {
    const int tap = tt >> 1, half = tt & 1;
    for (int t = tid; t < BN * 12; t += 256) {
      const int n = t / 12, ch = t % 12;
      const int off = ((tap * COUT + n0 + n) * COUT + half * 96 + ch * 8) * 2;
      async_ld16(&Bs[bufi][n][ch * 8], w2t, off);
    }
  };

  issueA(0, 0);
  issueB(0, 0);

  const int rowA = wm * 32 + (lane & 15);
  const int colB = wn * 32 + (lane & 15);
  int buf = 0;

  for (int tt = 0; tt < 18; ++tt) {
    async_wait0();
    __syncthreads();
    if (tt < 17) { issueA(tt + 1, buf ^ 1); issueB(tt + 1, buf ^ 1); }

#pragma unroll
    for (int kc = 0; kc < 96; kc += 32) {
      const bf16x16 a0 = lds_afrag(&As[buf][rowA][0],      kc, lane);
      const bf16x16 a1 = lds_afrag(&As[buf][rowA + 16][0], kc, lane);
      const bf16x16 b0 = lds_bfrag(&Bs[buf][colB][0],      kc, lane);
      const bf16x16 b1 = lds_bfrag(&Bs[buf][colB + 16][0], kc, lane);
      acc[0][0] = wmma_bf16(a0, b0, acc[0][0]);
      acc[0][1] = wmma_bf16(a0, b1, acc[0][1]);
      acc[1][0] = wmma_bf16(a1, b0, acc[1][0]);
      acc[1][1] = wmma_bf16(a1, b1, acc[1][1]);
    }
    buf ^= 1;
  }

  const float* bn2s = bnbuf + 2 * COUT;
  const float* bn2b = bnbuf + 3 * COUT;
  const int mhi = (lane & 16) ? 8 : 0;
#pragma unroll
  for (int j = 0; j < 2; ++j) {
    const int n = n0 + wn * 32 + j * 16 + (lane & 15);
    const float s2 = bn2s[n], o2 = bn2b[n];
    const float* cf = coef + (n % 3) * 12;
    float c[10];
#pragma unroll
    for (int q = 0; q < 10; ++q) c[q] = cf[q];
#pragma unroll
    for (int i = 0; i < 2; ++i) {
#pragma unroll
      for (int r = 0; r < 8; ++r) {
        const int m = m0 + wm * 32 + i * 16 + r + mhi;
        const float y = __builtin_fmaf(acc[i][j][r], s2, o2) + sc[m * COUT + n];
        const float a = rational_f(y, c);
        const int b = m / 784, rem = m % 784;
        out[(b * COUT + n) * 784 + rem] = a;   /* NCHW fp32 */
      }
    }
  }
}

/* ================= host launch ================= */

extern "C" void kernel_launch(void* const* d_in, const int* in_sizes, int n_in,
                              void* d_out, int out_size, void* d_ws, size_t ws_size,
                              hipStream_t stream) {
  (void)in_sizes; (void)n_in; (void)out_size; (void)ws_size;
  const float* x   = (const float*)d_in[0];
  const float* w1  = (const float*)d_in[1];
  const float* g1  = (const float*)d_in[2];
  const float* b1  = (const float*)d_in[3];
  const float* m1  = (const float*)d_in[4];
  const float* v1  = (const float*)d_in[5];
  const float* nr  = (const float*)d_in[6];
  const float* dr  = (const float*)d_in[7];
  const float* ng  = (const float*)d_in[8];
  const float* dg  = (const float*)d_in[9];
  const float* nb  = (const float*)d_in[10];
  const float* db  = (const float*)d_in[11];
  const float* w2  = (const float*)d_in[12];
  const float* g2  = (const float*)d_in[13];
  const float* b2  = (const float*)d_in[14];
  const float* m2  = (const float*)d_in[15];
  const float* v2  = (const float*)d_in[16];
  const float* wsk = (const float*)d_in[17];
  const float* gs  = (const float*)d_in[18];
  const float* bs  = (const float*)d_in[19];
  const float* ms  = (const float*)d_in[20];
  const float* vs  = (const float*)d_in[21];

  char* wsb = (char*)d_ws;
  __bf16* xb    = (__bf16*)(wsb + OFF_XB);
  __bf16* t1    = (__bf16*)(wsb + OFF_T1);
  float*  scb   = (float*)(wsb + OFF_SC);
  __bf16* w1t   = (__bf16*)(wsb + OFF_W1T);
  __bf16* w2t   = (__bf16*)(wsb + OFF_W2T);
  __bf16* wst   = (__bf16*)(wsb + OFF_WST);
  float*  bnbuf = (float*)(wsb + OFF_BN);
  float*  coef  = (float*)(wsb + OFF_COEF);

  prep_params<<<1, 256, 0, stream>>>(g1, b1, m1, v1, g2, b2, m2, v2, gs, bs, ms, vs,
                                     nr, dr, ng, dg, nb, db, bnbuf, coef);
  prep_x<<<dim3(HIN * HIN / 64, BB), 256, 0, stream>>>(x, xb);
  prep_w1<<<(9 * COUT * CIN + 255) / 256, 256, 0, stream>>>(w1, w1t);
  prep_w2<<<(9 * COUT * COUT + 255) / 256, 256, 0, stream>>>(w2, w2t);
  prep_wsk<<<(COUT * CIN + 255) / 256, 256, 0, stream>>>(wsk, wst);

  conv1_kernel<<<dim3(MTOT / BM, COUT / BN), 256, 0, stream>>>(xb, w1t, wst, bnbuf, coef, t1, scb);
  conv2_kernel<<<dim3(MTOT / BM, COUT / BN), 256, 0, stream>>>(t1, w2t, scb, bnbuf, coef,
                                                               (float*)d_out);
}
